// DIE_6597069766869
// MI455X (gfx1250) — matrix operand order repaired and checked by reading, not compiled, not run on hardware
//
#include <hip/hip_runtime.h>

// ---------------------------------------------------------------------------
// Per-pixel SE gate: out = sigmoid(W2 @ relu(W1 @ x)) * x   (B=8,C=512,HW=192^2)
//
// Roofline: 19.3 GFLOP vs 1.21 GB minimum HBM traffic => ~52us @ 23.3 TB/s.
// Staying memory-bound needs ~373 TFLOP/s of matmul => bf16 WMMA.
// All operands are staged ONCE into LDS in bf16, in layouts where every WMMA
// fragment (ISA 7.12.2 layouts) is exactly two contiguous 16B ds_load_b128:
//   x  -> [pixel][channel] (transposed, pitch 520)  : B fragments
//   W1 -> [row][k]   pitch 520                      : A fragments (GEMM1)
//   W2 -> [row][k]   pitch 40                       : A fragments (GEMM2)
//   h  -> [pixel][r] pitch 40                       : B fragment  (GEMM2)
// fp32 x for the final multiply is re-read from global (L2-hot: tile streamed
// moments earlier, concurrent working set << 192MB L2).
// ---------------------------------------------------------------------------

typedef __attribute__((ext_vector_type(16))) __bf16 v16bf;
typedef __attribute__((ext_vector_type(8)))  __bf16 v8bf;
typedef __attribute__((ext_vector_type(4)))  __bf16 v4bf;
typedef __attribute__((ext_vector_type(8)))  float  v8f;
typedef __attribute__((ext_vector_type(4)))  float  f32x4;

#define C_DIM   512
#define R_DIM   32
#define HW      (192 * 192)
#define B_DIM   8
#define TILE_P  128

// LDS pitches (bf16 elements); all padded off power-of-two to spread banks,
// all row starts / fragment offsets 16B aligned for ds_load_b128.
#define XT_PITCH 520
#define W1_PITCH 520
#define W2_PITCH 40
#define HT_PITCH 40

#define XT_OFF 0
#define W1_OFF (XT_OFF + TILE_P * XT_PITCH * 2)   // 133120
#define W2_OFF (W1_OFF + R_DIM  * W1_PITCH * 2)   // 166400
#define HT_OFF (W2_OFF + C_DIM  * W2_PITCH * 2)   // 207360
#define SMEM_BYTES (HT_OFF + TILE_P * HT_PITCH * 2)  // 217600 < 320KB WGP LDS

#define CAT16(lo, hi) __builtin_shufflevector(lo, hi, 0, 1, 2, 3, 4, 5, 6, 7, \
                                              8, 9, 10, 11, 12, 13, 14, 15)

__global__ __launch_bounds__(256, 1)
void die_se_wmma_kernel(const float* __restrict__ xg,
                        const float* __restrict__ w1,
                        const float* __restrict__ w2,
                        float* __restrict__ outg) {
    extern __shared__ char smem[];
    __bf16* xt  = (__bf16*)(smem + XT_OFF);
    __bf16* w1l = (__bf16*)(smem + W1_OFF);
    __bf16* w2l = (__bf16*)(smem + W2_OFF);
    __bf16* ht  = (__bf16*)(smem + HT_OFF);

    const int tid  = threadIdx.x;
    const int lane = tid & 31;
    const int wave = tid >> 5;      // 8 waves x 16-pixel N tiles
    const int ncol = lane & 15;
    const int hi   = lane >> 4;     // half-wave select (ISA WMMA layouts)

    const int tiles_per_b = HW / TILE_P;            // 288
    const int b  = blockIdx.x / tiles_per_b;
    const int p0 = (blockIdx.x % tiles_per_b) * TILE_P;

    const float* xb = xg   + (size_t)b * C_DIM * HW + p0;
    float*       ob = outg + (size_t)b * C_DIM * HW + p0;

    __builtin_prefetch(w1, 0, 0);
    __builtin_prefetch(w2, 0, 0);

    // ---- Stage W1 (32x512) -> bf16 LDS, row-major, padded pitch ----------
    for (int i = tid; i < (R_DIM * C_DIM) / 4; i += 256) {
        const int e = i << 2;
        const int r = e >> 9, k = e & 511;
        f32x4 v = *(const f32x4*)(w1 + r * C_DIM + k);
        v4bf o; o[0] = (__bf16)v[0]; o[1] = (__bf16)v[1];
                o[2] = (__bf16)v[2]; o[3] = (__bf16)v[3];
        *(v4bf*)(w1l + r * W1_PITCH + k) = o;
    }
    // ---- Stage W2 (512x32) -> bf16 LDS ------------------------------------
    for (int i = tid; i < (C_DIM * R_DIM) / 4; i += 256) {
        const int e = i << 2;
        const int r = e >> 5, k = e & 31;
        f32x4 v = *(const f32x4*)(w2 + r * R_DIM + k);
        v4bf o; o[0] = (__bf16)v[0]; o[1] = (__bf16)v[1];
                o[2] = (__bf16)v[2]; o[3] = (__bf16)v[3];
        *(v4bf*)(w2l + r * W2_PITCH + k) = o;
    }
    // ---- Stage x tile TRANSPOSED -> bf16 LDS [pixel][channel] -------------
    // coalesced: one wave instruction reads one 512B channel row
    for (int i = tid; i < C_DIM * (TILE_P / 4); i += 256) {
        const int c = i >> 5;
        const int j = (i & 31) << 2;
        f32x4 v = *(const f32x4*)(xb + (size_t)c * HW + j);
        xt[(j + 0) * XT_PITCH + c] = (__bf16)v[0];
        xt[(j + 1) * XT_PITCH + c] = (__bf16)v[1];
        xt[(j + 2) * XT_PITCH + c] = (__bf16)v[2];
        xt[(j + 3) * XT_PITCH + c] = (__bf16)v[3];
    }
    __syncthreads();

    const int pn   = wave * 16;
    const int pcol = pn + ncol;

    // ========== GEMM1: h = relu(W1 @ x)  (M=32, N=16/wave, K=512) ==========
    const __bf16* xrow = xt  + (size_t)pcol * XT_PITCH;   // B: this lane's pixel
    const __bf16* a0r  = w1l + (size_t)ncol * W1_PITCH;   // A: rows 0..15
    const __bf16* a1r  = a0r + (size_t)16   * W1_PITCH;   // A: rows 16..31

    v8f acc0 = {};
    v8f acc1 = {};
    for (int k0 = 0; k0 < C_DIM; k0 += 32) {
        // B fragment: b[0..7] = K k0+hi*8.., b[8..15] = K k0+16+hi*8..
        v8bf bl = *(const v8bf*)(xrow + k0 + hi * 8);
        v8bf bh8 = *(const v8bf*)(xrow + k0 + 16 + hi * 8);
        v16bf bx = CAT16(bl, bh8);
        // A fragment: a[0..7] = row[k0+hi*8 ..], a[8..15] = row[k0+16+hi*8 ..]
        const int kb = k0 + hi * 8;
        v16bf a0 = CAT16(*(const v8bf*)(a0r + kb), *(const v8bf*)(a0r + kb + 16));
        v16bf a1 = CAT16(*(const v8bf*)(a1r + kb), *(const v8bf*)(a1r + kb + 16));
        acc0 = __builtin_amdgcn_wmma_f32_16x16x32_bf16(false, a0, false, bx,
                                                       (short)0, acc0, false, false);
        acc1 = __builtin_amdgcn_wmma_f32_16x16x32_bf16(false, a1, false, bx,
                                                       (short)0, acc1, false, false);
    }

    // ReLU + store h transposed [pixel][r] (bf16)
    {
        __bf16* hrow = ht + (size_t)pcol * HT_PITCH;
        const int m = hi * 8;          // C-layout: lanes>=16 hold M+8
        #pragma unroll
        for (int r = 0; r < 8; ++r) {
            const float v0 = acc0[r] > 0.f ? acc0[r] : 0.f;
            const float v1 = acc1[r] > 0.f ? acc1[r] : 0.f;
            hrow[m + r]      = (__bf16)v0;
            hrow[16 + m + r] = (__bf16)v1;
        }
    }
    __syncthreads();

    // B fragment of h: bh[0..7] = K hi*8.., bh[8..15] = K 16+hi*8..
    v16bf bh;
    {
        const __bf16* hrow = ht + (size_t)pcol * HT_PITCH + hi * 8;
        bh = CAT16(*(const v8bf*)(hrow), *(const v8bf*)(hrow + 16));
    }

    // ==== GEMM2: gate = sigmoid(W2 @ h); out = gate * x  (M=512, K=32) =====
    for (int m0 = 0; m0 < C_DIM; m0 += 16) {
        const __bf16* wr = w2l + (size_t)(m0 + ncol) * W2_PITCH;
        const int kb = hi * 8;
        v16bf aw = CAT16(*(const v8bf*)(wr + kb), *(const v8bf*)(wr + kb + 16));

        v8f d = {};
        d = __builtin_amdgcn_wmma_f32_16x16x32_bf16(false, aw, false, bh,
                                                    (short)0, d, false, false);
        // C/D layout: VGPR r -> channel m0 + r + 8*hi, pixel pcol
        const int mb = m0 + hi * 8;
        #pragma unroll
        for (int r = 0; r < 8; ++r) {
            const int   cch = mb + r;
            const float g   = 1.0f / (1.0f + __expf(-d[r]));
            const float xv  = xb[(size_t)cch * HW + pcol];   // fp32, L2-hot
            ob[(size_t)cch * HW + pcol] = g * xv;
        }
    }
}

extern "C" void kernel_launch(void* const* d_in, const int* in_sizes, int n_in,
                              void* d_out, int out_size, void* d_ws, size_t ws_size,
                              hipStream_t stream) {
    const float* x  = (const float*)d_in[0];
    const float* w1 = (const float*)d_in[1];
    const float* w2 = (const float*)d_in[2];
    float* out = (float*)d_out;

    hipFuncSetAttribute(reinterpret_cast<const void*>(die_se_wmma_kernel),
                        hipFuncAttributeMaxDynamicSharedMemorySize, SMEM_BYTES);

    const dim3 grid(B_DIM * (HW / TILE_P));   // 2304 workgroups
    die_se_wmma_kernel<<<grid, 256, SMEM_BYTES, stream>>>(x, w1, w2, out);
}
